// TokenMixingMoE_69080253989464
// MI455X (gfx1250) — matrix operand order repaired and checked by reading, not compiled, not run on hardware
//
#include <hip/hip_runtime.h>
#include <hip/hip_bf16.h>
#include <math.h>

// ---------------------------------------------------------------------------
// TokenMixingMoE on MI455X (gfx1250), wave32 + WMMA bf16.
// B=4, NH=8, S=1024, H=128 -> N = 32768 tokens. E=8 experts, I=512, K=2.
// Dense evaluation of all experts weighted by the (mostly-zero) combine
// weights == reference semantics. All GEMM work through
// v_wmma_f32_16x16x32_bf16; LN/softmax/GELU in fp32.
// LDS row strides padded so stride/4 mod 64 != 0 (64-bank conflict-free
// fragment loads / tile stores).
// ---------------------------------------------------------------------------

typedef __attribute__((ext_vector_type(16))) __bf16 v16bf;
typedef __attribute__((ext_vector_type(8)))  __bf16 v8bf;
typedef __attribute__((ext_vector_type(8)))  float  v8f;

#define H_DIM 128
#define I_DIM 512
#define E_NUM 8
#define TOK   16
#define EPS_LN 1e-5f
#define INV_SQRT2 0.70710678118654752f

// padded LDS row strides (elements)
#define XPAD 132   // xbuf rows   (float): 132 % 64 = 4
#define A1P  136   // a1s rows    (bf16): 136/2=68 dwords -> 4-bank shift/row
#define TBP  516   // tbuf rows  (float): 516 % 64 = 4 -> 8g*4+... conflict-free
#define A2P  520   // a2s rows    (bf16): 520/2=260 dwords -> 4-bank shift/row

__device__ __forceinline__ float gelu_exact(float v) {
    return 0.5f * v * (1.0f + erff(v * INV_SQRT2));
}

// Convert fp32 expert weights to bf16 in workspace (L2-resident afterwards).
__global__ void moe_convert_weights(const float* __restrict__ w1,
                                    const float* __restrict__ w2,
                                    __bf16* __restrict__ w1b,
                                    __bf16* __restrict__ w2b,
                                    int n) {
    int idx = blockIdx.x * blockDim.x + threadIdx.x;
    if (idx < n) {
        w1b[idx] = (__bf16)w1[idx];
        w2b[idx] = (__bf16)w2[idx];
    }
}

__global__ __launch_bounds__(256)
void moe_tile_kernel(const float* __restrict__ x,
                     const float* __restrict__ gate_w,
                     const float* __restrict__ gate_b,
                     const float* __restrict__ ln1_g,
                     const float* __restrict__ ln1_b,
                     const float* __restrict__ ln2_g,
                     const float* __restrict__ ln2_b,
                     const float* __restrict__ b2,
                     const __bf16* __restrict__ w1b,   // [E][I][H] bf16
                     const __bf16* __restrict__ w2b,   // [E][H][I] bf16
                     float* __restrict__ out) {
    __shared__ float  xbuf[TOK][XPAD];       // x tile, then xn in-place
    __shared__ __bf16 a1s[TOK][A1P];         // gelu(ln1) bf16
    __shared__ float  tbuf[TOK][TBP];        // GEMM1 output fp32
    __shared__ __bf16 a2s[TOK][A2P];         // gelu(ln2) bf16
    __shared__ float  combine_s[TOK][E_NUM];
    __shared__ float  logits_s[TOK][E_NUM];
    __shared__ float  mu_s[TOK], rinv_s[TOK];

    const int tid  = threadIdx.x;
    const int tok0 = blockIdx.x * TOK;

    // ---- load x tile (coalesced) ----
    for (int idx = tid; idx < TOK * H_DIM; idx += 256)
        xbuf[idx >> 7][idx & (H_DIM - 1)] = x[tok0 * H_DIM + idx];
    __syncthreads();

    // ---- gate logits: 128 threads, one (token, expert) each ----
    if (tid < TOK * E_NUM) {
        int t = tid >> 3, e = tid & 7;
        float acc = gate_b[e];
        #pragma unroll 8
        for (int h = 0; h < H_DIM; ++h)
            acc += xbuf[t][h] * gate_w[e * H_DIM + h];
        logits_s[t][e] = acc;
    }
    __syncthreads();

    // ---- softmax + top-2 combine + LN1 stats (one thread per token) ----
    if (tid < TOK) {
        float m = -1e30f;
        for (int e = 0; e < E_NUM; ++e) m = fmaxf(m, logits_s[tid][e]);
        float p[E_NUM], s = 0.f;
        for (int e = 0; e < E_NUM; ++e) { p[e] = __expf(logits_s[tid][e] - m); s += p[e]; }
        float inv = 1.0f / s;
        float p1 = -1.f, p2 = -1.f; int i1 = -1, i2 = -1;
        for (int e = 0; e < E_NUM; ++e) {
            float pe = p[e] * inv;
            if (pe > p1)      { p2 = p1; i2 = i1; p1 = pe; i1 = e; }
            else if (pe > p2) { p2 = pe; i2 = e; }
        }
        for (int e = 0; e < E_NUM; ++e)
            combine_s[tid][e] = (e == i1) ? p1 : ((e == i2) ? p2 : 0.0f);

        float mu = 0.f;
        for (int h = 0; h < H_DIM; ++h) mu += xbuf[tid][h];
        mu *= (1.0f / H_DIM);
        float var = 0.f;
        for (int h = 0; h < H_DIM; ++h) { float d = xbuf[tid][h] - mu; var += d * d; }
        var *= (1.0f / H_DIM);
        mu_s[tid]   = mu;
        rinv_s[tid] = rsqrtf(var + EPS_LN);
    }
    __syncthreads();

    // ---- normalize x in place (expert-independent part of LN1) ----
    for (int idx = tid; idx < TOK * H_DIM; idx += 256) {
        int row = idx >> 7, h = idx & (H_DIM - 1);
        xbuf[row][h] = (xbuf[row][h] - mu_s[row]) * rinv_s[row];
    }
    __syncthreads();

    const int wave = tid >> 5;
    const int lane = tid & 31;
    const int l    = lane & 15;   // column / row-in-halfwave index
    const int g    = lane >> 4;   // half-wave group

    v8f oacc = {0.f, 0.f, 0.f, 0.f, 0.f, 0.f, 0.f, 0.f};

    for (int e = 0; e < E_NUM; ++e) {
        // ---- a1 = gelu(xn * ln1_g[e] + ln1_b[e]) -> bf16 ----
        for (int idx = tid; idx < TOK * H_DIM; idx += 256) {
            int row = idx >> 7, h = idx & (H_DIM - 1);
            float v = xbuf[row][h] * ln1_g[e * H_DIM + h] + ln1_b[e * H_DIM + h];
            a1s[row][h] = (__bf16)gelu_exact(v);
        }
        __syncthreads();

        // ---- GEMM1: t[16,512] = a1[16,128] @ w1[e][512,128]^T ----
        // A fragments (16x32 bf16): lane group g holds K = 16g..16g+15.
        v16bf afr[4];
        #pragma unroll
        for (int k = 0; k < 4; ++k) {
            int k0 = k * 32;
            v8bf c0 = *(const v8bf*)&a1s[l][k0 + 16 * g];
            v8bf c1 = *(const v8bf*)&a1s[l][k0 + 16 * g + 8];
            afr[k] = __builtin_shufflevector(c0, c1, 0, 1, 2, 3, 4, 5, 6, 7,
                                                     8, 9, 10, 11, 12, 13, 14, 15);
        }
        const __bf16* w1e = w1b + (size_t)e * I_DIM * H_DIM;
        #pragma unroll
        for (int nt = 0; nt < 4; ++nt) {
            int i0 = (wave * 4 + nt) * 16;
            v8f acc = {0.f, 0.f, 0.f, 0.f, 0.f, 0.f, 0.f, 0.f};
            #pragma unroll
            for (int k = 0; k < 4; ++k) {
                int k0 = k * 32;
                // B fragment (32x16 bf16): col = l, lane group g holds K = 16g..16g+15.
                v16bf bfr = *(const v16bf*)&w1e[(i0 + l) * H_DIM + k0 + 16 * g];
                acc = __builtin_amdgcn_wmma_f32_16x16x32_bf16(
                        false, afr[k], false, bfr, (short)0, acc, false, false);
            }
            #pragma unroll
            for (int r = 0; r < 8; ++r)
                tbuf[r + 8 * g][i0 + l] = acc[r];   // D: M = r + 8g, N = l
        }
        __syncthreads();

        // ---- LN2 + gelu -> a2 bf16 (each wave handles 2 of 16 rows) ----
        #pragma unroll
        for (int rr = 0; rr < 2; ++rr) {
            int row = wave * 2 + rr;
            float s0 = 0.f, s1 = 0.f;
            #pragma unroll
            for (int j = 0; j < 16; ++j) {
                float v = tbuf[row][lane + 32 * j];
                s0 += v; s1 += v * v;
            }
            #pragma unroll
            for (int off = 16; off > 0; off >>= 1) {
                s0 += __shfl_xor(s0, off, 32);
                s1 += __shfl_xor(s1, off, 32);
            }
            float mu   = s0 * (1.0f / I_DIM);
            float var  = s1 * (1.0f / I_DIM) - mu * mu;
            float rinv = rsqrtf(var + EPS_LN);
            #pragma unroll
            for (int j = 0; j < 16; ++j) {
                int i = lane + 32 * j;
                float v = (tbuf[row][i] - mu) * rinv * ln2_g[e * I_DIM + i]
                          + ln2_b[e * I_DIM + i];
                a2s[row][i] = (__bf16)gelu_exact(v);
            }
        }
        __syncthreads();

        // ---- GEMM2: out_e[16,128] = a2[16,512] @ w2[e][128,512]^T ----
        // Two independent accumulation chains (even/odd k) to relax the
        // serial WMMA D->C dependency.
        const int h0 = wave * 16;
        const __bf16* w2e = w2b + (size_t)e * H_DIM * I_DIM;
        v8f d0 = {0.f, 0.f, 0.f, 0.f, 0.f, 0.f, 0.f, 0.f};
        v8f d1 = {0.f, 0.f, 0.f, 0.f, 0.f, 0.f, 0.f, 0.f};
        #pragma unroll
        for (int k = 0; k < 16; k += 2) {
            {
                int k0 = k * 32;
                v8bf c0 = *(const v8bf*)&a2s[l][k0 + 16 * g];
                v8bf c1 = *(const v8bf*)&a2s[l][k0 + 16 * g + 8];
                v16bf af = __builtin_shufflevector(c0, c1, 0, 1, 2, 3, 4, 5, 6, 7,
                                                           8, 9, 10, 11, 12, 13, 14, 15);
                v16bf bf = *(const v16bf*)&w2e[(h0 + l) * I_DIM + k0 + 16 * g];
                d0 = __builtin_amdgcn_wmma_f32_16x16x32_bf16(
                        false, af, false, bf, (short)0, d0, false, false);
            }
            {
                int k0 = (k + 1) * 32;
                v8bf c0 = *(const v8bf*)&a2s[l][k0 + 16 * g];
                v8bf c1 = *(const v8bf*)&a2s[l][k0 + 16 * g + 8];
                v16bf af = __builtin_shufflevector(c0, c1, 0, 1, 2, 3, 4, 5, 6, 7,
                                                           8, 9, 10, 11, 12, 13, 14, 15);
                v16bf bf = *(const v16bf*)&w2e[(h0 + l) * I_DIM + k0 + 16 * g];
                d1 = __builtin_amdgcn_wmma_f32_16x16x32_bf16(
                        false, af, false, bf, (short)0, d1, false, false);
            }
        }
        float b2v = b2[e * H_DIM + h0 + l];
        #pragma unroll
        for (int r = 0; r < 8; ++r) {
            float cw = combine_s[r + 8 * g][e];
            oacc[r] += cw * (d0[r] + d1[r] + b2v);
        }
        __syncthreads();
    }

    // ---- write out tile ----
    const int h0 = wave * 16;
    #pragma unroll
    for (int r = 0; r < 8; ++r)
        out[(size_t)(tok0 + r + 8 * g) * H_DIM + h0 + l] = oacc[r];
}

extern "C" void kernel_launch(void* const* d_in, const int* in_sizes, int n_in,
                              void* d_out, int out_size, void* d_ws, size_t ws_size,
                              hipStream_t stream) {
    const float* x      = (const float*)d_in[0];
    const float* gate_w = (const float*)d_in[1];
    const float* gate_b = (const float*)d_in[2];
    const float* ln1_g  = (const float*)d_in[3];
    const float* ln1_b  = (const float*)d_in[4];
    const float* w1     = (const float*)d_in[5];
    const float* ln2_g  = (const float*)d_in[6];
    const float* ln2_b  = (const float*)d_in[7];
    const float* w2     = (const float*)d_in[8];
    const float* b2     = (const float*)d_in[9];
    float* out = (float*)d_out;

    const int n_w = E_NUM * I_DIM * H_DIM;       // 524288 elements each
    __bf16* w1bf = (__bf16*)d_ws;
    __bf16* w2bf = w1bf + n_w;                   // total 2 MB of workspace

    moe_convert_weights<<<(n_w + 255) / 256, 256, 0, stream>>>(w1, w2, w1bf, w2bf, n_w);

    const int n_tokens = out_size / H_DIM;       // 32768
    moe_tile_kernel<<<n_tokens / TOK, 256, 0, stream>>>(
        x, gate_w, gate_b, ln1_g, ln1_b, ln2_g, ln2_b, b2, w1bf, w2bf, out);
}
